// BlockAttentionResidual_3178275799214
// MI455X (gfx1250) — matrix operand order repaired and checked by reading, not compiled, not run on hardware
//
#include <hip/hip_runtime.h>
#include <hip/hip_bf16.h>
#include <math.h>

// ---------------------------------------------------------------------------
// MI455X (gfx1250) transformer block.
//  - All GEMMs (QKV / O / UP / DOWN) run on v_wmma_f32_16x16x32_f16.
//  - GEMM A-tiles staged global->LDS by the Tensor Data Mover (TDM) with
//    hardware LDS padding; synchronized with s_wait_tensorcnt.
//  - Block-sparse causal attention (4 blocks of 512) as a flash-attention
//    kernel, QK^T and P*V both on WMMA.
// Shapes: B=2, S=2048, HIDDEN=2048, HEADS=32, HD=64, INTER=4096.
// ---------------------------------------------------------------------------

typedef __attribute__((ext_vector_type(16))) _Float16     v16h;
typedef __attribute__((ext_vector_type(8)))  _Float16     v8h;
typedef __attribute__((ext_vector_type(8)))  float        v8f;
typedef __attribute__((ext_vector_type(4)))  unsigned int v4u;
typedef __attribute__((ext_vector_type(8)))  int          v8i;
typedef __attribute__((ext_vector_type(4)))  int          v4i;

#define WMMA_F16(a, b, c) \
  __builtin_amdgcn_wmma_f32_16x16x32_f16(false, (a), false, (b), (short)0, (c), false, false)

#if defined(__has_builtin)
#if __has_builtin(__builtin_amdgcn_tensor_load_to_lds) && \
    __has_builtin(__builtin_amdgcn_s_wait_tensorcnt)
#define USE_TDM 1
#endif
#endif
#ifndef USE_TDM
#define USE_TDM 0
#endif

static constexpr int HIDDEN = 2048;
static constexpr int HEADS  = 32;
static constexpr int HD     = 64;
static constexpr int BLK    = 512;   // attention block size (S=2048 / 4 blocks)

// ===========================================================================
// GEMM: C[M,N] = A[M,K] * W[N,K]^T   (A,W f16 row-major; C f32; optional
// residual add).  Tile 128x128x32; 256 threads = 8 wave32 waves laid out
// 4(M) x 2(N); each wave owns a 32x64 patch = 2x4 WMMA accumulators.
// A tile is DMA'd by the TDM (wave 0 issues; TENSORcnt synchronized); the
// TDM pad feature (16 DWORDs data + 4 DWORDs pad) produces the 40-f16 LDS
// row stride directly.  B tile is transposed into LDS by the loader threads.
// ===========================================================================
static constexpr int BM = 128, BN = 128, BK = 32;
static constexpr int LDSA = BK + 8;   // 40 f16 row stride (80B = 64B + 16B pad)
static constexpr int LDSB = BN + 8;   // 136 f16 row stride (272B, 16B aligned)

__global__ __launch_bounds__(256)
void gemm_f16_wmma_kernel(const _Float16* __restrict__ A,
                          const _Float16* __restrict__ W,
                          float* __restrict__ C,
                          const float* __restrict__ R,   // residual or null
                          int M, int N, int K)
{
  __shared__ _Float16 As[2][BM][LDSA];     // [m][k]
  __shared__ _Float16 Bs[2][BK][LDSB];     // [k][n]  (transposed on load)

  const int tid  = threadIdx.x;
  const int lane = tid & 31;
  const int w    = tid >> 5;       // 0..7
  const int wm   = w >> 1;         // 0..3 -> 32-row slab
  const int wn   = w & 1;          // 0..1 -> 64-col slab
  const int m0   = blockIdx.y * BM;
  const int n0   = blockIdx.x * BN;

  const int lrow = tid >> 1;            // 0..127
  const int lcol = (tid & 1) * 16;      // 0 / 16

  v8f acc[2][4];
#pragma unroll
  for (int i = 0; i < 2; ++i)
#pragma unroll
    for (int j = 0; j < 4; ++j) {
      v8f z = {0.f, 0.f, 0.f, 0.f, 0.f, 0.f, 0.f, 0.f};
      acc[i][j] = z;
    }

  const int nk = K / BK;

#if USE_TDM
  // ---- Tensor DMA descriptor (D#) for one 128x32 f16 A tile -------------
  // group1 is loop-invariant; group0's global_addr advances by BK*2 bytes.
  const unsigned ldsA[2] = {
    (unsigned)(unsigned long long)(&As[0][0][0]),
    (unsigned)(unsigned long long)(&As[1][0][0]),
  };
  v8i g1;
  g1[0] = (1 << 16)          // data_size = 2 bytes
        | (1 << 20)          // pad_enable
        | (3 << 22)          // pad_interval: 16 DWORDs (= 64B tile row)
        | (3 << 25);         // pad_amount:   4 DWORDs (= 16B -> 80B stride)
  g1[1] = (int)((unsigned)K << 16);                           // tensor_dim0 lo16
  g1[2] = (int)((((unsigned)K >> 16) & 0xffffu)
        | ((unsigned)M << 16));                               // dim0 hi | dim1 lo
  g1[3] = (int)((((unsigned)M >> 16) & 0xffffu)
        | ((unsigned)BK << 16));                              // dim1 hi | tile_dim0
  g1[4] = BM;                                                 // tile_dim1 | tile_dim2=0
  g1[5] = (int)(unsigned)K;                                   // tensor_dim0_stride lo32
  g1[6] = 0;                                                  // stride hi | dim1_stride lo
  g1[7] = 0;
  const v4i gz4 = {0, 0, 0, 0};
  const v8i gz8 = {0, 0, 0, 0, 0, 0, 0, 0};

  auto tdm_load_A = [&](int kt, int buf) {
    unsigned long long ga =
        (unsigned long long)(const void*)(A + (size_t)m0 * K + (size_t)kt * BK);
    v4u g0;
    g0[0] = 1u;                                               // count=1, user mode
    g0[1] = ldsA[buf];                                        // lds_addr (bytes)
    g0[2] = (unsigned)(ga & 0xffffffffu);                     // global_addr lo
    g0[3] = (unsigned)((ga >> 32) & 0x01ffffffu) | (2u << 30);// addr hi | type=2
    __builtin_amdgcn_tensor_load_to_lds(g0, g1, gz4, gz4, gz8, 0);
  };
#else
  auto load_A_tile = [&](int kt, int buf) {
    const _Float16* ag = A + (size_t)(m0 + lrow) * K + kt * BK + lcol;
    __builtin_prefetch(ag + BK, 0, 1);                 // global_prefetch_b8
    v8h a0 = *(const v8h*)ag;
    v8h a1 = *(const v8h*)(ag + 8);
    *(v8h*)&As[buf][lrow][lcol]     = a0;
    *(v8h*)&As[buf][lrow][lcol + 8] = a1;
  };
#endif

  auto load_B_tile = [&](int kt, int buf) {
    const _Float16* bg = W + (size_t)(n0 + lrow) * K + kt * BK + lcol;
    __builtin_prefetch(bg + BK, 0, 1);                 // global_prefetch_b8
    v8h b0 = *(const v8h*)bg;
    v8h b1 = *(const v8h*)(bg + 8);
#pragma unroll
    for (int i = 0; i < 8; ++i) Bs[buf][lcol + i][lrow]     = b0[i];
#pragma unroll
    for (int i = 0; i < 8; ++i) Bs[buf][lcol + 8 + i][lrow] = b1[i];
  };

#if USE_TDM
  if (w == 0) tdm_load_A(0, 0);
#else
  load_A_tile(0, 0);
#endif
  load_B_tile(0, 0);
#if USE_TDM
  if (w == 0) __builtin_amdgcn_s_wait_tensorcnt(0);
#endif
  __syncthreads();

  const int mlo = lane & 15;   // A-frag: row within 16
  const int kg  = lane >> 4;   // A-frag: K half-group selector (ISA 7.12.2)

  for (int kt = 0; kt < nk; ++kt) {
    const int buf = kt & 1;
    if (kt + 1 < nk) {
#if USE_TDM
      if (w == 0) tdm_load_A(kt + 1, buf ^ 1);
#else
      load_A_tile(kt + 1, buf ^ 1);
#endif
      load_B_tile(kt + 1, buf ^ 1);
    }

    v16h af[2];
#pragma unroll
    for (int mt = 0; mt < 2; ++mt) {
      const _Float16* p = &As[buf][wm * 32 + mt * 16 + mlo][0];
      v8h lo = *(const v8h*)(p + kg * 8);        // K = kg*8 .. kg*8+7
      v8h hi = *(const v8h*)(p + 16 + kg * 8);   // K = 16+kg*8 .. +7
      v16h a;
#pragma unroll
      for (int i = 0; i < 8; ++i) { a[i] = lo[i]; a[i + 8] = hi[i]; }
      af[mt] = a;
    }

    v16h bf[4];
#pragma unroll
    for (int nt = 0; nt < 4; ++nt) {
      // B-frag: lane%16 = N, elements = K in table order (kg*8+i, 16+kg*8+i)
      const _Float16* p = &Bs[buf][0][wn * 64 + nt * 16 + mlo];
      v16h b;
#pragma unroll
      for (int i = 0; i < 8; ++i) { b[i] = p[(kg * 8 + i) * LDSB]; b[i + 8] = p[(16 + kg * 8 + i) * LDSB]; }
      bf[nt] = b;
    }

#pragma unroll
    for (int mt = 0; mt < 2; ++mt)
#pragma unroll
      for (int nt = 0; nt < 4; ++nt)
        acc[mt][nt] = WMMA_F16(af[mt], bf[nt], acc[mt][nt]);

#if USE_TDM
    if (kt + 1 < nk && w == 0) __builtin_amdgcn_s_wait_tensorcnt(0);
#endif
    __syncthreads();
  }

  // Epilogue: C/D layout -> m = r + 8*(lane>>4), n = lane&15.
  // The residual null-check is block-uniform: hoist it so both bodies are
  // straight-line; strength-reduce the index (idx += N per row).
  const int hl = lane >> 4;
  const int nl = lane & 15;
  if (R != nullptr) {
#pragma unroll
    for (int mt = 0; mt < 2; ++mt)
#pragma unroll
      for (int nt = 0; nt < 4; ++nt) {
        size_t idx = (size_t)(m0 + wm * 32 + mt * 16 + 8 * hl) * N
                   + (size_t)(n0 + wn * 64 + nt * 16 + nl);
#pragma unroll
        for (int r = 0; r < 8; ++r, idx += N)
          C[idx] = acc[mt][nt][r] + R[idx];
      }
  } else {
#pragma unroll
    for (int mt = 0; mt < 2; ++mt)
#pragma unroll
      for (int nt = 0; nt < 4; ++nt) {
        size_t idx = (size_t)(m0 + wm * 32 + mt * 16 + 8 * hl) * N
                   + (size_t)(n0 + wn * 64 + nt * 16 + nl);
#pragma unroll
        for (int r = 0; r < 8; ++r, idx += N)
          C[idx] = acc[mt][nt][r];
      }
  }
}

// ===========================================================================
// RMSNorm (fp32 in, f16 out), one 256-thread block per row of 2048.
// ===========================================================================
__global__ __launch_bounds__(256)
void rmsnorm_f16_kernel(const float* __restrict__ X,
                        const float* __restrict__ Wt,
                        _Float16* __restrict__ O)
{
  __shared__ float red[256];
  const int row = blockIdx.x;
  const float* xr = X + (size_t)row * HIDDEN;
  float ss = 0.f;
  for (int i = threadIdx.x; i < HIDDEN; i += 256) { float v = xr[i]; ss += v * v; }
  red[threadIdx.x] = ss;
  __syncthreads();
  for (int s = 128; s > 0; s >>= 1) {
    if (threadIdx.x < s) red[threadIdx.x] += red[threadIdx.x + s];
    __syncthreads();
  }
  const float scale = rsqrtf(red[0] / (float)HIDDEN + 1e-5f);
  for (int i = threadIdx.x; i < HIDDEN; i += 256)
    O[(size_t)row * HIDDEN + i] = (_Float16)(xr[i] * scale * Wt[i]);
}

// ===========================================================================
// fp32 -> f16 flat convert (weights).
// ===========================================================================
__global__ __launch_bounds__(256)
void f32_to_f16_kernel(const float* __restrict__ in, _Float16* __restrict__ out,
                       size_t n)
{
  size_t i = (size_t)blockIdx.x * 256 + threadIdx.x;
  if (i < n) out[i] = (_Float16)in[i];
}

// ===========================================================================
// RoPE + QKV split: qkv f32 [B*S, 6144] -> Q/K/V f16 [B][H][S][D].
// One thread per (row, h*64+d).
// ===========================================================================
__global__ __launch_bounds__(256)
void rope_split_kernel(const float* __restrict__ qkv,
                       _Float16* __restrict__ Q,
                       _Float16* __restrict__ Kd,
                       _Float16* __restrict__ Vd,
                       int S)
{
  const size_t idx = (size_t)blockIdx.x * 256 + threadIdx.x;  // B*S*2048 total
  const int    col = (int)(idx & (HIDDEN - 1));
  const size_t row = idx >> 11;
  const int s = (int)(row % S);
  const int b = (int)(row / S);
  const int h = col >> 6;
  const int d = col & 63;

  const float invf = __powf(10000.f, -(float)(d & 31) / 32.f);
  const float ang  = (float)s * invf;
  float si, c;
  __sincosf(ang, &si, &c);

  const float* rp = qkv + row * (3 * HIDDEN);
  const float qv = rp[col],          qp = rp[col ^ 32];
  const float kv = rp[HIDDEN + col], kp = rp[HIDDEN + (col ^ 32)];
  const float sgn = (d < 32) ? -1.f : 1.f;

  const size_t o = ((size_t)(b * HEADS + h) * S + s) * HD + d;
  Q[o]  = (_Float16)(qv * c + sgn * qp * si);
  Kd[o] = (_Float16)(kv * c + sgn * kp * si);
  Vd[o] = (_Float16)rp[2 * HIDDEN + col];
}

// ===========================================================================
// Flash attention, block-sparse causal (512-wide blocks).
// grid = (S/128, HEADS, B), 256 threads; wave w owns q rows q0..q0+15.
// All waves iterate the full 512-key block uniformly (causal mask zeros the
// invalid part), so K/V chunk staging uses plain workgroup barriers.
// ===========================================================================
__global__ __launch_bounds__(256)
void flash_attn_kernel(const _Float16* __restrict__ Qm,
                       const _Float16* __restrict__ Km,
                       const _Float16* __restrict__ Vm,
                       _Float16* __restrict__ ctx,
                       int S)
{
  __shared__ _Float16 Kls[HD][32 + 8];       // K chunk transposed: [d][key]
  __shared__ _Float16 Vls[32][HD + 8];       // V chunk: [key][d]
  __shared__ _Float16 Pls[8][16][32 + 8];    // per-wave P re-layout scratch

  const int tid  = threadIdx.x;
  const int lane = tid & 31;
  const int w    = tid >> 5;
  const int h    = blockIdx.y;
  const int b    = blockIdx.z;
  const int q0   = blockIdx.x * 128 + w * 16;

  const size_t base = (size_t)(b * HEADS + h) * S * HD;
  const _Float16* Qp = Qm + base;
  const _Float16* Kp = Km + base;
  const _Float16* Vp = Vm + base;

  const int mlo = lane & 15;
  const int kg  = lane >> 4;
  const int hl  = lane >> 4;
  const int nl  = lane & 15;

  // Q A-fragments for the two 32-wide d-chunks, pre-scaled by 1/sqrt(64).
  v16h qf[2];
#pragma unroll
  for (int c = 0; c < 2; ++c) {
    const _Float16* qrow = Qp + (size_t)(q0 + mlo) * HD + c * 32;
    v8h lo = *(const v8h*)(qrow + kg * 8);
    v8h hi = *(const v8h*)(qrow + 16 + kg * 8);
    v16h a;
#pragma unroll
    for (int i = 0; i < 8; ++i) {
      a[i]     = (_Float16)((float)lo[i] * 0.125f);
      a[i + 8] = (_Float16)((float)hi[i] * 0.125f);
    }
    qf[c] = a;
  }

  v8f accd[4];
#pragma unroll
  for (int dt = 0; dt < 4; ++dt) {
    v8f z = {0.f, 0.f, 0.f, 0.f, 0.f, 0.f, 0.f, 0.f};
    accd[dt] = z;
  }
  float rmax[8], rsum[8];
#pragma unroll
  for (int r = 0; r < 8; ++r) { rmax[r] = -3.0e38f; rsum[r] = 0.f; }

  const int kstart = (q0 / BLK) * BLK;   // uniform: 128 divides 512

  for (int kc = 0; kc < BLK; kc += 32) {
    const int key0 = kstart + kc;
    __syncthreads();
    {   // cooperative staging of 32 keys x 64 dims of K (transposed) and V
      const int keyl = tid >> 3;          // 0..31
      const int dc   = (tid & 7) * 8;     // 0..56
      v8h kv = *(const v8h*)(Kp + (size_t)(key0 + keyl) * HD + dc);
#pragma unroll
      for (int i = 0; i < 8; ++i) Kls[dc + i][keyl] = kv[i];
      v8h vv = *(const v8h*)(Vp + (size_t)(key0 + keyl) * HD + dc);
      *(v8h*)&Vls[keyl][dc] = vv;
    }
    __syncthreads();

    // S = Q K^T over this 32-key chunk (two 16-key n-tiles)
    v8f sf[2];
    {
      v8f z = {0.f, 0.f, 0.f, 0.f, 0.f, 0.f, 0.f, 0.f};
      sf[0] = z; sf[1] = z;
    }
#pragma unroll
    for (int c = 0; c < 2; ++c)
#pragma unroll
      for (int nt = 0; nt < 2; ++nt) {
        const _Float16* p = &Kls[c * 32][nt * 16 + nl];
        v16h bb;
#pragma unroll
        for (int i = 0; i < 8; ++i) { bb[i] = p[(kg * 8 + i) * 40]; bb[i + 8] = p[(16 + kg * 8 + i) * 40]; }
        sf[nt] = WMMA_F16(qf[c], bb, sf[nt]);
      }

    // Online softmax (rows of C-frag: m = r + 8*hl, col = nl / nl+16)
    float p0[8], p1[8];
#pragma unroll
    for (int r = 0; r < 8; ++r) {
      const int qrow = q0 + 8 * hl + r;
      const int k0n  = key0 + nl;
      const bool v0  = (k0n <= qrow);
      const bool v1  = (k0n + 16 <= qrow);
      float s0 = v0 ? sf[0][r] : -3.0e38f;
      float s1 = v1 ? sf[1][r] : -3.0e38f;
      float mx = fmaxf(s0, s1);
      for (int off = 1; off < 16; off <<= 1)
        mx = fmaxf(mx, __shfl_xor(mx, off, 32));
      const float nm   = fmaxf(rmax[r], mx);
      const float corr = __expf(rmax[r] - nm);
      rmax[r] = nm;
      const float e0 = v0 ? __expf(s0 - nm) : 0.f;
      const float e1 = v1 ? __expf(s1 - nm) : 0.f;
      float ps = e0 + e1;
      for (int off = 1; off < 16; off <<= 1)
        ps += __shfl_xor(ps, off, 32);
      rsum[r] = rsum[r] * corr + ps;
      p0[r] = e0; p1[r] = e1;
#pragma unroll
      for (int dt = 0; dt < 4; ++dt) accd[dt][r] *= corr;
    }

    // Re-layout P (C-frag) -> A-frag via per-wave LDS scratch
    __builtin_amdgcn_wave_barrier();
#pragma unroll
    for (int r = 0; r < 8; ++r) {
      Pls[w][8 * hl + r][nl]      = (_Float16)p0[r];
      Pls[w][8 * hl + r][16 + nl] = (_Float16)p1[r];
    }
    __builtin_amdgcn_wave_barrier();
    v16h pf;
    {
      const _Float16* pp = &Pls[w][mlo][0];
      v8h lo = *(const v8h*)(pp + kg * 8);
      v8h hi = *(const v8h*)(pp + 16 + kg * 8);
#pragma unroll
      for (int i = 0; i < 8; ++i) { pf[i] = lo[i]; pf[i + 8] = hi[i]; }
    }

    // context += P * V  (B-frag: lane%16 = d, elements = keys in table order)
#pragma unroll
    for (int dt = 0; dt < 4; ++dt) {
      const _Float16* vp = &Vls[0][dt * 16 + nl];
      v16h vf;
#pragma unroll
      for (int i = 0; i < 8; ++i) { vf[i] = vp[(kg * 8 + i) * (HD + 8)]; vf[i + 8] = vp[(16 + kg * 8 + i) * (HD + 8)]; }
      accd[dt] = WMMA_F16(pf, vf, accd[dt]);
    }
  }

  // Normalize and write context f16 at [(b*S + q), h*64 + d]
#pragma unroll
  for (int dt = 0; dt < 4; ++dt)
#pragma unroll
    for (int r = 0; r < 8; ++r) {
      const int qrow = q0 + 8 * hl + r;
      const int d    = dt * 16 + nl;
      const float ov = accd[dt][r] / rsum[r];
      ctx[(size_t)(b * S + qrow) * HIDDEN + h * HD + d] = (_Float16)ov;
    }
}

// ===========================================================================
// SwiGLU: up f32 [rows, 8192] -> act f16 [rows, 4096]
// ===========================================================================
__global__ __launch_bounds__(256)
void silu_mul_kernel(const float* __restrict__ up, _Float16* __restrict__ out)
{
  const size_t idx = (size_t)blockIdx.x * 256 + threadIdx.x;  // rows*4096
  const size_t row = idx >> 12;
  const int    col = (int)(idx & 4095);
  const float g = up[row * 8192 + col];
  const float v = up[row * 8192 + 4096 + col];
  out[idx] = (_Float16)((g / (1.f + __expf(-g))) * v);
}

// ===========================================================================
// Host-side orchestration
// ===========================================================================
extern "C" void kernel_launch(void* const* d_in, const int* in_sizes, int n_in,
                              void* d_out, int out_size, void* d_ws, size_t ws_size,
                              hipStream_t stream)
{
  const float* x       = (const float*)d_in[0];
  const float* attn_w  = (const float*)d_in[1];
  const float* ffn_w   = (const float*)d_in[2];
  const float* w_qkv   = (const float*)d_in[3];
  const float* w_o     = (const float*)d_in[4];
  const float* w_up    = (const float*)d_in[5];
  const float* w_down  = (const float*)d_in[6];
  float* out           = (float*)d_out;

  const int S  = 2048;
  const int BS = in_sizes[0] / HIDDEN;        // B*S rows (= 4096)
  const int B  = BS / S;

  // Workspace carving (256B aligned)
  char*  ws  = (char*)d_ws;
  size_t off = 0;
  auto carve = [&](size_t bytes) -> char* {
    char* p = ws + off;
    off += (bytes + 255) & ~(size_t)255;
    return p;
  };
  _Float16* h16     = (_Float16*)carve((size_t)BS * HIDDEN * 2);
  _Float16* wqkv16  = (_Float16*)carve((size_t)3 * HIDDEN * HIDDEN * 2);
  _Float16* wo16    = (_Float16*)carve((size_t)HIDDEN * HIDDEN * 2);
  _Float16* wup16   = (_Float16*)carve((size_t)4 * HIDDEN * HIDDEN * 2);
  _Float16* wdown16 = (_Float16*)carve((size_t)2 * HIDDEN * HIDDEN * 2);
  float*    qkv32   = (float*)carve((size_t)BS * 3 * HIDDEN * 4);
  _Float16* q16     = (_Float16*)carve((size_t)BS * HIDDEN * 2);
  _Float16* k16     = (_Float16*)carve((size_t)BS * HIDDEN * 2);
  _Float16* v16     = (_Float16*)carve((size_t)BS * HIDDEN * 2);
  _Float16* ctx16   = (_Float16*)carve((size_t)BS * HIDDEN * 2);
  float*    xattn   = (float*)carve((size_t)BS * HIDDEN * 4);
  _Float16* h2_16   = (_Float16*)carve((size_t)BS * HIDDEN * 2);
  float*    up32    = (float*)carve((size_t)BS * 4 * HIDDEN * 4);
  _Float16* act16   = (_Float16*)carve((size_t)BS * 2 * HIDDEN * 2);

  auto cvt = [&](const float* src, _Float16* dst, size_t n) {
    f32_to_f16_kernel<<<(unsigned)((n + 255) / 256), 256, 0, stream>>>(src, dst, n);
  };
  cvt(w_qkv,  wqkv16,  (size_t)3 * HIDDEN * HIDDEN);
  cvt(w_o,    wo16,    (size_t)HIDDEN * HIDDEN);
  cvt(w_up,   wup16,   (size_t)4 * HIDDEN * HIDDEN);
  cvt(w_down, wdown16, (size_t)2 * HIDDEN * HIDDEN);

  auto gemm = [&](const _Float16* A, const _Float16* Wm, float* C,
                  const float* R, int M, int N, int K) {
    dim3 grid(N / BN, M / BM);
    gemm_f16_wmma_kernel<<<grid, 256, 0, stream>>>(A, Wm, C, R, M, N, K);
  };

  // 1) h = rmsnorm(x)
  rmsnorm_f16_kernel<<<BS, 256, 0, stream>>>(x, attn_w, h16);
  // 2) qkv = h @ w_qkv^T
  gemm(h16, wqkv16, qkv32, nullptr, BS, 3 * HIDDEN, HIDDEN);
  // 3) rope + split into (B,H,S,D) f16
  rope_split_kernel<<<(unsigned)(((size_t)BS * HIDDEN) / 256), 256, 0, stream>>>(
      qkv32, q16, k16, v16, S);
  // 4) block-sparse causal flash attention
  {
    dim3 grid(S / 128, HEADS, B);
    flash_attn_kernel<<<grid, 256, 0, stream>>>(q16, k16, v16, ctx16, S);
  }
  // 5) x_attn = x + ctx @ w_o^T
  gemm(ctx16, wo16, xattn, x, BS, HIDDEN, HIDDEN);
  // 6) h2 = rmsnorm(x_attn)
  rmsnorm_f16_kernel<<<BS, 256, 0, stream>>>(xattn, ffn_w, h2_16);
  // 7) up = h2 @ w_up^T
  gemm(h2_16, wup16, up32, nullptr, BS, 4 * HIDDEN, HIDDEN);
  // 8) act = silu(gate) * value
  silu_mul_kernel<<<(unsigned)(((size_t)BS * 2 * HIDDEN) / 256), 256, 0, stream>>>(
      up32, act16);
  // 9) out = x_attn + act @ w_down^T
  gemm(act16, wdown16, out, xattn, BS, HIDDEN, 2 * HIDDEN);
}
